// WaveLayer_36850819399878
// MI455X (gfx1250) — matrix operand order repaired and checked by reading, not compiled, not run on hardware
//
#include <hip/hip_runtime.h>

// ---------------------------------------------------------------------------
// Types for CDNA5 WMMA
// ---------------------------------------------------------------------------
typedef __attribute__((ext_vector_type(16))) __bf16 v16bf;
typedef __attribute__((ext_vector_type(8)))  float  v8f;

union FragBF { uint4 q[2]; v16bf v; };

__device__ __forceinline__ unsigned short f2bf(float f) {
  unsigned u = __float_as_uint(f);
  u += 0x7FFFu + ((u >> 16) & 1u);   // round-to-nearest-even
  return (unsigned short)(u >> 16);
}

#define FLAG_RELU   1
#define FLAG_ATOMIC 2

// ---------------------------------------------------------------------------
// Generic bf16 WMMA GEMM: C = A[M,K] @ B[K,N] (row-major bf16 in, f32 acc).
// Block tile 128x128, BK=32; 8 waves in a 4x2 grid, each wave 32x64 =
// 2x4 v_wmma_f32_16x16x32_bf16 accumulators.
// Double-buffered LDS; A tiles move global->LDS via GLOBAL_LOAD_ASYNC_TO_LDS_B128
// (ASYNCcnt), B tiles global->regs->LDS (transposed) overlapped with WMMA.
// Assumes K % 8 == 0 and Nn % 8 == 0 (true for every launch here); zero-pads
// whole 8-element chunks at the K tail / M tail.
// Batched via gridDim.z = batches*splitK; split-K partials atomicAdd into Cf.
// ---------------------------------------------------------------------------
__global__ __launch_bounds__(256)
void gemm_bf16_wmma(const unsigned short* __restrict__ A, int lda, long long sA,
                    const unsigned short* __restrict__ B, int ldb, long long sB,
                    float* Cf, int ldc, long long sC,
                    unsigned short* Cb, int ldcb, long long sCb,
                    const float* bias, long long sBias,
                    const float* add, int ldadd, long long sAdd,
                    int M, int Nn, int K, int splitK, int kchunk, int flags)
{
  __shared__ __align__(16) unsigned int lds_a[2][128 * 20]; // 128 rows x 32 bf16 (+pad)
  __shared__ __align__(16) unsigned int lds_b[2][128 * 20]; // 128 cols x 32 bf16 (B^T)

  const int tid   = threadIdx.x;
  const int batch = blockIdx.z / splitK;
  const int split = blockIdx.z % splitK;
  A += (long long)batch * sA;
  B += (long long)batch * sB;
  if (Cf)   Cf   += (long long)batch * sC;
  if (Cb)   Cb   += (long long)batch * sCb;
  if (bias) bias += (long long)batch * sBias;
  if (add)  add  += (long long)batch * sAdd;

  const int mBase = blockIdx.x * 128;
  const int nBase = blockIdx.y * 128;
  const int kBeg  = split * kchunk;
  const int kEnd  = (K < kBeg + kchunk) ? K : (kBeg + kchunk);
  const int nT    = (kEnd > kBeg) ? ((kEnd - kBeg + 31) >> 5) : 0;

  const int lane = tid & 31;
  const int wv   = tid >> 5;
  const int wm   = wv & 3;        // wave row 0..3  (32 rows each)
  const int wn   = wv >> 2;       // wave col 0..1  (64 cols each)
  const int half = lane >> 4;     // 0/1
  const int lr   = lane & 15;

  v8f acc[2][4];
  const v8f vzero = {0.f,0.f,0.f,0.f,0.f,0.f,0.f,0.f};
#pragma unroll
  for (int i = 0; i < 2; ++i)
#pragma unroll
    for (int j = 0; j < 4; ++j) acc[i][j] = vzero;

  // --- stage A tile kt into lds_a[buf] via async global->LDS DMA ---
  auto issueA = [&](int kt, int buf) {
#pragma unroll
    for (int it = 0; it < 2; ++it) {
      int c = tid + it * 256;
      int row = c >> 2, kc = c & 3;
      int gm = mBase + row, gk = kt + kc * 8;
      unsigned int* ldst = &lds_a[buf][row * 20 + kc * 4];
      if (gm < M && gk + 8 <= kEnd) {
        const unsigned short* p = A + (long long)gm * lda + gk;
        unsigned ldsOff = (unsigned)(size_t)ldst;      // low 32 bits = LDS byte addr
        asm volatile("global_load_async_to_lds_b128 %0, %1, off"
                     :: "v"(ldsOff), "v"(p) : "memory");
      } else {
        *(uint4*)ldst = make_uint4(0u, 0u, 0u, 0u);    // zero padding (DScnt path)
      }
    }
  };
  // --- load B tile kt global -> regs ---
  auto loadB = [&](int kt, uint4* regs) {
#pragma unroll
    for (int it = 0; it < 2; ++it) {
      int c = tid + it * 256;
      int krow = c >> 4, nc = (c & 15) << 3;
      int gk = kt + krow, gn = nBase + nc;
      uint4 v4 = make_uint4(0u, 0u, 0u, 0u);
      if (gk < kEnd && gn + 8 <= Nn)
        v4 = *(const uint4*)(B + (long long)gk * ldb + gn);
      regs[it] = v4;
    }
  };
  // --- store B regs transposed into lds_b[buf] : Bs[n][k], pitch 40 ushorts ---
  auto storeB = [&](const uint4* regs, int buf) {
    unsigned short* Bus = (unsigned short*)lds_b[buf];
#pragma unroll
    for (int it = 0; it < 2; ++it) {
      int c = tid + it * 256;
      int krow = c >> 4, nc = (c & 15) << 3;
      unsigned wb[4] = {regs[it].x, regs[it].y, regs[it].z, regs[it].w};
#pragma unroll
      for (int j = 0; j < 8; ++j)
        Bus[(nc + j) * 40 + krow] = (unsigned short)(wb[j >> 1] >> ((j & 1) * 16));
    }
  };
  // --- fragments & 8 WMMA from lds buffers ---
  auto compute = [&](int buf) {
    FragBF a[2], b[4];
#pragma unroll
    for (int sm = 0; sm < 2; ++sm) {
      int r = wm * 32 + sm * 16 + lr;
      a[sm].q[0] = *(const uint4*)&lds_a[buf][r * 20 + half * 4];     // K 0..7 / 8..15
      a[sm].q[1] = *(const uint4*)&lds_a[buf][r * 20 + 8 + half * 4]; // K 16..23 / 24..31
    }
#pragma unroll
    for (int sn = 0; sn < 4; ++sn) {
      int n = wn * 64 + sn * 16 + lr;
      b[sn].q[0] = *(const uint4*)&lds_b[buf][n * 20 + half * 4];     // K 0..7 / 8..15
      b[sn].q[1] = *(const uint4*)&lds_b[buf][n * 20 + 8 + half * 4]; // K 16..23 / 24..31
    }
#pragma unroll
    for (int sm = 0; sm < 2; ++sm)
#pragma unroll
      for (int sn = 0; sn < 4; ++sn)
        acc[sm][sn] = __builtin_amdgcn_wmma_f32_16x16x32_bf16(
            false, a[sm].v, false, b[sn].v, (short)0, acc[sm][sn], false, false);
  };

  // --- software-pipelined main loop (parity-unrolled, one barrier/iter) ---
  uint4 rEven[2], rOdd[2];
  if (nT > 0) { issueA(kBeg, 0); loadB(kBeg, rEven); }
  int i = 0;
  while (i < nT) {
    // even tile -> buffers 0
    storeB(rEven, 0);
    asm volatile("s_wait_asynccnt 0x0" ::: "memory");
    __syncthreads();
    if (i + 1 < nT) { issueA(kBeg + (i + 1) * 32, 1); loadB(kBeg + (i + 1) * 32, rOdd); }
    compute(0);
    ++i; if (i >= nT) break;
    // odd tile -> buffers 1
    storeB(rOdd, 1);
    asm volatile("s_wait_asynccnt 0x0" ::: "memory");
    __syncthreads();
    if (i + 1 < nT) { issueA(kBeg + (i + 1) * 32, 0); loadB(kBeg + (i + 1) * 32, rEven); }
    compute(1);
    ++i;
  }

  // ---- epilogue ----
#pragma unroll
  for (int sm = 0; sm < 2; ++sm) {
#pragma unroll
    for (int sn = 0; sn < 4; ++sn) {
      int gn = nBase + wn * 64 + sn * 16 + lr;
#pragma unroll
      for (int i2 = 0; i2 < 8; ++i2) {
        int gm = mBase + wm * 32 + sm * 16 + half * 8 + i2;
        if (gm >= M || gn >= Nn) continue;
        float v = acc[sm][sn][i2];
        if (flags & FLAG_ATOMIC) {
          atomicAdd(&Cf[(long long)gm * ldc + gn], v);
        } else {
          if (bias) v += bias[gn];
          if (add)  v += add[(long long)gm * ldadd + gn];
          if (flags & FLAG_RELU) v = v > 0.f ? v : 0.f;
          if (Cf) Cf[(long long)gm * ldc + gn]  = v;
          if (Cb) Cb[(long long)gm * ldcb + gn] = f2bf(v);
        }
      }
    }
  }
}

// ---------------------------------------------------------------------------
// Elementwise / utility kernels
// ---------------------------------------------------------------------------
__global__ void cvt_bf16(const float* __restrict__ s, unsigned short* __restrict__ d,
                         long long n) {
  long long i = blockIdx.x * (long long)blockDim.x + threadIdx.x;
  if (i < n) d[i] = f2bf(s[i]);
}

__global__ void transpose_bf16(const float* __restrict__ s, unsigned short* __restrict__ dT,
                               int n, int k) { // s:[n,k] -> dT:[k,n]
  long long i = blockIdx.x * (long long)blockDim.x + threadIdx.x;
  long long tot = (long long)n * k;
  if (i < tot) {
    long long kk = i / n, nn = i % n;
    dT[i] = f2bf(s[nn * (long long)k + kk]);
  }
}

__global__ void build_sig(const float* __restrict__ fs, float* __restrict__ sig,
                          int k, int tj) { // fs:[k,tj] -> sig:[tj,k]
  int i = blockIdx.x * blockDim.x + threadIdx.x;
  if (i < k * tj) { int t = i / k, kk = i % k; sig[i] = fs[kk * tj + t]; }
}

__global__ void fill_f32(float* p, float v, long long n) {
  long long i = blockIdx.x * (long long)blockDim.x + threadIdx.x;
  if (i < n) p[i] = v;
}

__global__ void scale_cvt(const float* __restrict__ u, const float* __restrict__ sig,
                          unsigned short* __restrict__ ub, int k, int s, int tj) {
  long long i = blockIdx.x * (long long)blockDim.x + threadIdx.x;
  long long tot = (long long)tj * k * s;
  if (i < tot) {
    int t  = (int)(i / ((long long)k * s));
    int kk = (int)((i / s) % k);
    ub[i] = f2bf(u[i] * sig[t * k + kk]);
  }
}

__global__ void deg_count(const int* __restrict__ dst, float* deg, long long e) {
  long long i = blockIdx.x * (long long)blockDim.x + threadIdx.x;
  if (i < e) atomicAdd(&deg[dst[i]], 1.0f);
}

__global__ void to_rsqrt(float* p, long long n) {
  long long i = blockIdx.x * (long long)blockDim.x + threadIdx.x;
  if (i < n) p[i] = rsqrtf(p[i]);
}

__global__ void gcn_selfinit(const float* __restrict__ xw, const float* __restrict__ dinv,
                             const float* __restrict__ b, float* __restrict__ h,
                             long long nd, int d) {
  long long i = blockIdx.x * (long long)blockDim.x + threadIdx.x;
  if (i < nd) {
    long long r = i / d; int c = (int)(i % d);
    float di = dinv[r];
    h[i] = di * di * xw[i] + b[c];
  }
}

__global__ void gcn_scatter(const int* __restrict__ src, const int* __restrict__ dst,
                            const float* __restrict__ dinv, const float* __restrict__ xw,
                            float* __restrict__ hloc, long long e, int d) {
  long long t = blockIdx.x * (long long)blockDim.x + threadIdx.x;
  long long eid = t >> 7;            // 128 lanes/edge, 4 floats/lane (d==512)
  int lane = (int)(t & 127);
  if (eid >= e) return;
  int s = src[eid], dd = dst[eid];
  float w = dinv[s] * dinv[dd];
  const float4 v = *(const float4*)(xw + (long long)s * d + lane * 4);
  float* o = hloc + (long long)dd * d + lane * 4;
  atomicAdd(o + 0, w * v.x); atomicAdd(o + 1, w * v.y);
  atomicAdd(o + 2, w * v.z); atomicAdd(o + 3, w * v.w);
}

// Per-column sum / sumsq of (p1 + p2) over rows (256 threads: cols tid, tid+256)
__global__ __launch_bounds__(256)
void bn_stats(const float* __restrict__ p1, const float* __restrict__ p2,
              int n, int d, float* sum, float* sumsq) {
  int c0 = threadIdx.x, c1 = threadIdx.x + 256;
  long long r0 = (long long)blockIdx.x * 64;
  long long rE = r0 + 64 < n ? r0 + 64 : n;
  float s0 = 0.f, q0 = 0.f, s1 = 0.f, q1 = 0.f;
  for (long long r = r0; r < rE; ++r) {
    const float* a = p1 + r * d;
    const float* b = p2 ? p2 + r * d : nullptr;
    float x = a[c0] + (b ? b[c0] : 0.f); s0 += x; q0 += x * x;
    if (c1 < d) { float y = a[c1] + (b ? b[c1] : 0.f); s1 += y; q1 += y * y; }
  }
  atomicAdd(&sum[c0], s0); atomicAdd(&sumsq[c0], q0);
  if (c1 < d) { atomicAdd(&sum[c1], s1); atomicAdd(&sumsq[c1], q1); }
}

__global__ void bn_norm(const float* __restrict__ p1, const float* __restrict__ p2,
                        const float* __restrict__ sum, const float* __restrict__ sumsq,
                        const float* __restrict__ g, const float* __restrict__ b,
                        float* __restrict__ dst, unsigned short* __restrict__ bfout,
                        long long nd, int d, float invN, int addFlag) {
  long long i = blockIdx.x * (long long)blockDim.x + threadIdx.x;
  if (i >= nd) return;
  int c = (int)(i % d);
  float mean = sum[c] * invN;
  float var  = sumsq[c] * invN - mean * mean;
  float v = p1[i] + (p2 ? p2[i] : 0.f);
  v = (v - mean) * rsqrtf(var + 1e-5f) * g[c] + b[c];
  if (addFlag) v += dst[i];
  dst[i] = v;
  if (bfout) bfout[i] = f2bf(v);
}

// ---------------------------------------------------------------------------
// Host launcher
// ---------------------------------------------------------------------------
extern "C" void kernel_launch(void* const* d_in, const int* in_sizes, int n_in,
                              void* d_out, int out_size, void* d_ws, size_t ws_size,
                              hipStream_t stream) {
  (void)n_in; (void)out_size; (void)ws_size;
  const int N = 50000, D = 512, Kc = 128, TJ = 4, S = 128, D2 = 1024;
  const long long ND = (long long)N * D, NK = (long long)N * Kc, N2D = (long long)N * D2;
  const long long E = in_sizes[1] / 2;

  const float* x    = (const float*)d_in[0];
  const int*   esrc = (const int*)d_in[1];
  const int*   edst = esrc + E;
  const float* evc  = (const float*)d_in[2];
  const float* fs   = (const float*)d_in[3];
  const float* gcnW = (const float*)d_in[4];
  const float* gcnb = (const float*)d_in[5];
  const float* linW = (const float*)d_in[6];
  const float* linb = (const float*)d_in[7];
  const float* fusW = (const float*)d_in[8];
  const float* fusb = (const float*)d_in[9];
  const float* bn1lg = (const float*)d_in[10], *bn1lb = (const float*)d_in[11];
  const float* bn1ag = (const float*)d_in[12], *bn1ab = (const float*)d_in[13];
  const float* bn2g  = (const float*)d_in[14], *bn2b  = (const float*)d_in[15];
  const float* ff1W  = (const float*)d_in[16], *ff1b  = (const float*)d_in[17];
  const float* ff2W  = (const float*)d_in[18], *ff2b  = (const float*)d_in[19];
  float* out = (float*)d_out;

  char* ws = (char*)d_ws;
  size_t off = 0;
  auto alloc = [&](size_t bytes) -> void* {
    void* p = ws + off; off = (off + bytes + 255) & ~(size_t)255; return p;
  };
  unsigned short* x_bf    = (unsigned short*)alloc(ND * 2);
  unsigned short* evc_bf  = (unsigned short*)alloc(NK * 2);
  unsigned short* evcT_bf = (unsigned short*)alloc(NK * 2);
  unsigned short* gcnW_bf = (unsigned short*)alloc((size_t)D * D * 2);
  unsigned short* linW_bf = (unsigned short*)alloc((size_t)TJ * D * S * 2);
  unsigned short* fusW_bf = (unsigned short*)alloc((size_t)D * D * 2);
  unsigned short* ff1W_bf = (unsigned short*)alloc((size_t)D * D2 * 2);
  unsigned short* ff2W_bf = (unsigned short*)alloc((size_t)D2 * D * 2);
  float*          sig     = (float*)alloc((size_t)TJ * Kc * 4);
  float*          u_f     = (float*)alloc((size_t)TJ * Kc * S * 4);
  unsigned short* u_bf    = (unsigned short*)alloc((size_t)TJ * Kc * S * 2);
  float*          u2_f    = (float*)alloc((size_t)TJ * Kc * S * 4);
  unsigned short* u2_bf   = (unsigned short*)alloc((size_t)TJ * Kc * S * 2);
  float*          dinv    = (float*)alloc((size_t)N * 4);
  float*          bsum    = (float*)alloc((size_t)D * 4);
  float*          bsq     = (float*)alloc((size_t)D * 4);
  float*          bufA    = (float*)alloc(ND * 4);   // xw -> h_attn
  float*          bufB    = (float*)alloc(ND * 4);   // h_local -> ff2 out (BN2 input)
  float*          bufC    = (float*)alloc(ND * 4);   // h combined
  unsigned short* bfA     = (unsigned short*)alloc(ND * 2);  // ht -> comb -> h_bf16
  unsigned short* bfB     = (unsigned short*)alloc(ND * 2);  // v
  unsigned short* bfF     = (unsigned short*)alloc(N2D * 2); // ff1 out

  dim3 blk(256);
  auto gs = [](long long n) { return dim3((unsigned)((n + 255) / 256)); };
  const dim3 gMxD((N + 127) / 128, D / 128, 1);     // M=N, Nn=512
  const dim3 gMxS((N + 127) / 128, 1, TJ);          // M=N, Nn=128, batched
  const dim3 gSplit(1, 1, TJ * 63);                 // spectral projections
  const dim3 gMx2D((N + 127) / 128, D2 / 128, 1);   // M=N, Nn=1024

  // ---- conversions ----
  cvt_bf16<<<gs(ND), blk, 0, stream>>>(x, x_bf, ND);
  cvt_bf16<<<gs(NK), blk, 0, stream>>>(evc, evc_bf, NK);
  transpose_bf16<<<gs(NK), blk, 0, stream>>>(evc, evcT_bf, N, Kc);
  cvt_bf16<<<gs((long long)D * D), blk, 0, stream>>>(gcnW, gcnW_bf, (long long)D * D);
  cvt_bf16<<<gs((long long)TJ * D * S), blk, 0, stream>>>(linW, linW_bf, (long long)TJ * D * S);
  cvt_bf16<<<gs((long long)D * D), blk, 0, stream>>>(fusW, fusW_bf, (long long)D * D);
  cvt_bf16<<<gs((long long)D * D2), blk, 0, stream>>>(ff1W, ff1W_bf, (long long)D * D2);
  cvt_bf16<<<gs((long long)D2 * D), blk, 0, stream>>>(ff2W, ff2W_bf, (long long)D2 * D);
  build_sig<<<gs(TJ * Kc), blk, 0, stream>>>(fs, sig, Kc, TJ);

  // ---- GCN branch: xw = x @ gcn_W ----
  gemm_bf16_wmma<<<gMxD, blk, 0, stream>>>(
      x_bf, D, 0, gcnW_bf, D, 0,
      bufA, D, 0, nullptr, 0, 0,
      nullptr, 0, nullptr, 0, 0,
      N, D, D, 1, D, 0);
  fill_f32<<<gs(N), blk, 0, stream>>>(dinv, 1.0f, N);          // deg = 1 (self loop)
  deg_count<<<gs(E), blk, 0, stream>>>(edst, dinv, E);
  to_rsqrt<<<gs(N), blk, 0, stream>>>(dinv, N);
  gcn_selfinit<<<gs(ND), blk, 0, stream>>>(bufA, dinv, gcnb, bufB, ND, D);
  gcn_scatter<<<gs(E * 128), blk, 0, stream>>>(esrc, edst, dinv, bufA, bufB, E, D);

  // ---- Wave branch: ht[t] = x @ lin_W[t] + lin_b[t]  (bf16 out) ----
  gemm_bf16_wmma<<<gMxS, blk, 0, stream>>>(
      x_bf, D, 0, linW_bf, S, (long long)D * S,
      nullptr, 0, 0, bfA, S, (long long)N * S,
      linb, S, nullptr, 0, 0,
      N, S, D, 1, D, 0);
  // u[t] = evc^T @ ht[t]   (split-K atomic, K = 50000)
  fill_f32<<<gs((long long)TJ * Kc * S), blk, 0, stream>>>(u_f, 0.f, (long long)TJ * Kc * S);
  gemm_bf16_wmma<<<gSplit, blk, 0, stream>>>(
      evcT_bf, N, 0, bfA, S, (long long)N * S,
      u_f, S, (long long)Kc * S, nullptr, 0, 0,
      nullptr, 0, nullptr, 0, 0,
      Kc, S, N, 63, 800, FLAG_ATOMIC);
  scale_cvt<<<gs((long long)TJ * Kc * S), blk, 0, stream>>>(u_f, sig, u_bf, Kc, S, TJ);
  // v[t] = relu(evc @ u[t])  (bf16)
  gemm_bf16_wmma<<<gMxS, blk, 0, stream>>>(
      evc_bf, Kc, 0, u_bf, S, (long long)Kc * S,
      nullptr, 0, 0, bfB, S, (long long)N * S,
      nullptr, 0, nullptr, 0, 0,
      N, S, Kc, 1, Kc, FLAG_RELU);
  // u2[t] = evc^T @ v[t]
  fill_f32<<<gs((long long)TJ * Kc * S), blk, 0, stream>>>(u2_f, 0.f, (long long)TJ * Kc * S);
  gemm_bf16_wmma<<<gSplit, blk, 0, stream>>>(
      evcT_bf, N, 0, bfB, S, (long long)N * S,
      u2_f, S, (long long)Kc * S, nullptr, 0, 0,
      nullptr, 0, nullptr, 0, 0,
      Kc, S, N, 63, 800, FLAG_ATOMIC);
  scale_cvt<<<gs((long long)TJ * Kc * S), blk, 0, stream>>>(u2_f, sig, u2_bf, Kc, S, TJ);
  // comb[n, t*S+s] = (evc @ u2[t])[n,s]  (interleaved store via ldcb=D, batch stride S)
  gemm_bf16_wmma<<<gMxS, blk, 0, stream>>>(
      evc_bf, Kc, 0, u2_bf, S, (long long)Kc * S,
      nullptr, 0, 0, bfA, D, (long long)S,
      nullptr, 0, nullptr, 0, 0,
      N, S, Kc, 1, Kc, 0);
  // h_attn = relu(comb @ fusion_W + fusion_b)  (f32)
  gemm_bf16_wmma<<<gMxD, blk, 0, stream>>>(
      bfA, D, 0, fusW_bf, D, 0,
      bufA, D, 0, nullptr, 0, 0,
      fusb, 0, nullptr, 0, 0,
      N, D, D, 1, D, FLAG_RELU);

  // ---- BN1l over (x + h_local) -> bufC ----
  fill_f32<<<gs(D), blk, 0, stream>>>(bsum, 0.f, D);
  fill_f32<<<gs(D), blk, 0, stream>>>(bsq, 0.f, D);
  bn_stats<<<dim3((N + 63) / 64), blk, 0, stream>>>(x, bufB, N, D, bsum, bsq);
  bn_norm<<<gs(ND), blk, 0, stream>>>(x, bufB, bsum, bsq, bn1lg, bn1lb,
                                      bufC, nullptr, ND, D, 1.0f / N, 0);
  // ---- BN1a over (x + h_attn), add into bufC, emit h as bf16 ----
  fill_f32<<<gs(D), blk, 0, stream>>>(bsum, 0.f, D);
  fill_f32<<<gs(D), blk, 0, stream>>>(bsq, 0.f, D);
  bn_stats<<<dim3((N + 63) / 64), blk, 0, stream>>>(x, bufA, N, D, bsum, bsq);
  bn_norm<<<gs(ND), blk, 0, stream>>>(x, bufA, bsum, bsq, bn1ag, bn1ab,
                                      bufC, bfA, ND, D, 1.0f / N, 1);

  // ---- FF block ----
  gemm_bf16_wmma<<<gMx2D, blk, 0, stream>>>(
      bfA, D, 0, ff1W_bf, D2, 0,
      nullptr, 0, 0, bfF, D2, 0,
      ff1b, 0, nullptr, 0, 0,
      N, D2, D, 1, D, FLAG_RELU);
  gemm_bf16_wmma<<<gMxD, blk, 0, stream>>>(
      bfF, D2, 0, ff2W_bf, D, 0,
      bufB, D, 0, nullptr, 0, 0,
      ff2b, 0, bufC, D, 0,
      N, D, D2, 1, D2, 0);

  // ---- BN2 over (h + ff) -> out ----
  fill_f32<<<gs(D), blk, 0, stream>>>(bsum, 0.f, D);
  fill_f32<<<gs(D), blk, 0, stream>>>(bsq, 0.f, D);
  bn_stats<<<dim3((N + 63) / 64), blk, 0, stream>>>(bufB, nullptr, N, D, bsum, bsq);
  bn_norm<<<gs(ND), blk, 0, stream>>>(bufB, nullptr, bsum, bsq, bn2g, bn2b,
                                      out, nullptr, ND, D, 1.0f / N, 0);
}
